// TSMamba_41042707480767
// MI455X (gfx1250) — compile-verified, matrix operand order checked by reading, not run on hardware
//
#include <hip/hip_runtime.h>
#include <hip/hip_bf16.h>
#include <stdint.h>

// Problem constants (match reference)
#define BB      4
#define LL      4096
#define IN_DIM  32
#define DM      256      // D_MODEL
#define DI      512      // D_INNER
#define DST     16       // D_STATE
#define DTR     16       // DT_RANK
#define NPROJ   48       // DT_RANK + 2*D_STATE
#define NLAYERS 2
#define M_ROWS  (BB*LL)  // 16384
#define OUT_PER_B 128    // OUTPUT_DIM*HORIZON

typedef unsigned short u16;
typedef __attribute__((ext_vector_type(8)))  __bf16 v8bf;
typedef __attribute__((ext_vector_type(16))) __bf16 v16bf;
typedef __attribute__((ext_vector_type(8)))  float  v8f;

__device__ __forceinline__ u16 f2bf(float f) {
  union { float f; uint32_t u; } c; c.f = f;
  uint32_t r = c.u + 0x7FFFu + ((c.u >> 16) & 1u);   // round-to-nearest-even
  return (u16)(r >> 16);
}

// ---------------- f32 -> bf16 conversion ----------------
__global__ void k_f32_to_bf16(const float* __restrict__ s, u16* __restrict__ d, int n) {
  int i = blockIdx.x * blockDim.x + threadIdx.x;
  if (i < n) d[i] = f2bf(s[i]);
}

// ---------------- WMMA bf16 GEMM: C[M,N] = A[M,K] * W[N,K]^T (+bias) ----------------
// 16-bit A fragment layout (CDNA5 ISA): lanes 0-15 hold row M=lane with
// K = {0..7, 16..23}; lanes 16-31 hold row M=lane-16 with K = {8..15, 24..31}.
// B operand = column n of B = row n of W, same per-lane K striping.
__device__ __forceinline__ v16bf load_frag(const u16* __restrict__ base, int stride,
                                           int row, int kb, int lane) {
  int half = lane >> 4;
  int r    = lane & 15;
  const u16* p = base + (size_t)(row + r) * stride + kb + half * 8;
  v8bf lo = *(const v8bf*)(const void*)p;         // K offsets 0..7   (+8*half)
  v8bf hi = *(const v8bf*)(const void*)(p + 16);  // K offsets 16..23 (+8*half)
  return __builtin_shufflevector(lo, hi, 0,1,2,3,4,5,6,7,8,9,10,11,12,13,14,15);
}

// Each wave computes a 64x16 output strip (4 M-tiles x 1 N-tile): the weight
// (B) fragment is loaded once per K-step and reused in registers across the
// 4 WMMAs -> 2.5 b128 loads per WMMA instead of 4.
__global__ void k_gemm_bf16(const u16* __restrict__ A, const u16* __restrict__ W,
                            const float* __restrict__ bias, float* __restrict__ C,
                            int M, int N, int K, int accumulate) {
  int wavesPerBlk = blockDim.x >> 5;
  int wave = blockIdx.x * wavesPerBlk + (threadIdx.x >> 5);
  int lane = threadIdx.x & 31;
  int Nt = N >> 4;
  int total = (M >> 6) * Nt;                 // M strips of 64
  if (wave >= total) return;                 // wave-uniform branch: EXEC stays all-1s
  int ms = wave / Nt;                        // M-strip index
  int nt = wave - ms * Nt;                   // consecutive waves share the A strip (cache)
  int mrow = ms << 6;                        // base row of the 64-row strip
  int nrow = nt << 4;

  v8f acc0 = {0.f,0.f,0.f,0.f,0.f,0.f,0.f,0.f};
  v8f acc1 = acc0, acc2 = acc0, acc3 = acc0;
  for (int kb = 0; kb < K; kb += 32) {
    v16bf b  = load_frag(W, K, nrow, kb, lane);
    v16bf a0 = load_frag(A, K, mrow +  0, kb, lane);
    v16bf a1 = load_frag(A, K, mrow + 16, kb, lane);
    v16bf a2 = load_frag(A, K, mrow + 32, kb, lane);
    v16bf a3 = load_frag(A, K, mrow + 48, kb, lane);
    acc0 = __builtin_amdgcn_wmma_f32_16x16x32_bf16(false, a0, false, b, (short)0, acc0, false, false);
    acc1 = __builtin_amdgcn_wmma_f32_16x16x32_bf16(false, a1, false, b, (short)0, acc1, false, false);
    acc2 = __builtin_amdgcn_wmma_f32_16x16x32_bf16(false, a2, false, b, (short)0, acc2, false, false);
    acc3 = __builtin_amdgcn_wmma_f32_16x16x32_bf16(false, a3, false, b, (short)0, acc3, false, false);
  }
  // C/D layout: VGPR i, lanes 0-15 -> M=i, lanes 16-31 -> M=i+8; N = lane&15
  int half = lane >> 4;
  int col  = nrow + (lane & 15);
  float bv = bias ? bias[col] : 0.f;
  v8f accs[4] = {acc0, acc1, acc2, acc3};
  #pragma unroll
  for (int tmi = 0; tmi < 4; tmi++) {
    int rbase = mrow + (tmi << 4) + (half << 3);
    #pragma unroll
    for (int i = 0; i < 8; i++) {
      size_t idx = (size_t)(rbase + i) * N + col;
      float v = accs[tmi][i] + bv;
      C[idx] = accumulate ? (C[idx] + v) : v;
    }
  }
}

// ---------------- causal depthwise conv (k=4) + SiLU, emits f32 + bf16 ----------------
__global__ void k_conv_silu(const float* __restrict__ xz, const float* __restrict__ cw,
                            const float* __restrict__ cb, float* __restrict__ xi,
                            u16* __restrict__ xibf) {
  size_t t = (size_t)blockIdx.x * blockDim.x + threadIdx.x;
  if (t >= (size_t)M_ROWS * DI) return;
  int d = (int)(t & (DI - 1));
  size_t m = t >> 9;                 // row index in [0, B*L)
  int l = (int)(m & (LL - 1));       // position within its sequence
  float acc = cb[d];
  #pragma unroll
  for (int k = 0; k < 4; k++) {
    int ls = l + k - 3;
    if (ls >= 0) {
      size_t mm = m + (size_t)k - 3;
      acc += xz[mm * (size_t)(2 * DI) + d] * cw[d * 4 + k];
    }
  }
  float s = acc / (1.f + __expf(-acc));     // silu
  xi[t] = s;
  xibf[t] = f2bf(s);
}

// ---------------- dt = softplus(dbl[:, :16] @ dt_w^T + dt_b), f32 VALU (K=16) ----------
__global__ void k_dtproj(const float* __restrict__ dbl, const float* __restrict__ dtw,
                         const float* __restrict__ dtb, float* __restrict__ dt) {
  size_t t = (size_t)blockIdx.x * blockDim.x + threadIdx.x;
  if (t >= (size_t)M_ROWS * DI) return;
  int d = (int)(t & (DI - 1));
  size_t m = t >> 9;
  const float* r = dbl + m * (size_t)NPROJ;
  const float* w = dtw + d * DTR;
  float acc = dtb[d];
  #pragma unroll
  for (int j = 0; j < DTR; j++) acc = __builtin_fmaf(r[j], w[j], acc);
  dt[t] = (acc > 20.f) ? acc : log1pf(expf(acc));
}

// ---------------- selective scan: 16 blocks x 128 channels, sequential in L ----------
__global__ void k_scan(const float* __restrict__ dbl, const float* __restrict__ dt,
                       const float* __restrict__ xi, const float* __restrict__ xz,
                       const float* __restrict__ A_log, const float* __restrict__ Dp,
                       u16* __restrict__ ybf) {
  int b = blockIdx.x >> 2;
  int d = ((blockIdx.x & 3) << 7) + threadIdx.x;
  float A[DST], h[DST];
  #pragma unroll
  for (int s = 0; s < DST; s++) { A[s] = -__expf(A_log[d * DST + s]); h[s] = 0.f; }
  float Dv = Dp[d];
  size_t base = (size_t)b * LL;
  for (int l = 0; l < LL; l++) {
    size_t m = base + l;
    const float* dr = dbl + m * (size_t)NPROJ;
    float Bv[DST], Cv[DST];
    #pragma unroll
    for (int q = 0; q < 4; q++) {
      *(float4*)&Bv[4 * q] = *(const float4*)(dr + DTR + 4 * q);
      *(float4*)&Cv[4 * q] = *(const float4*)(dr + DTR + DST + 4 * q);
    }
    float dtv = dt[m * DI + d];
    float xv  = xi[m * DI + d];
    float dtx = dtv * xv;
    float y = 0.f;
    #pragma unroll
    for (int s = 0; s < DST; s++) {
      float dA = __expf(dtv * A[s]);
      h[s] = __builtin_fmaf(dA, h[s], dtx * Bv[s]);
      y = __builtin_fmaf(h[s], Cv[s], y);
    }
    y = __builtin_fmaf(xv, Dv, y);                       // + xi * Dp
    float r = xz[m * (size_t)(2 * DI) + DI + d];         // res branch
    y *= r / (1.f + __expf(-r));                         // * silu(res)
    ybf[m * DI + d] = f2bf(y);
  }
}

// ---------------- final LayerNorm (last token only) + head ----------------
__global__ void k_ln_head(const float* __restrict__ h, const float* __restrict__ g,
                          const float* __restrict__ bta, const float* __restrict__ hw,
                          const float* __restrict__ hb, float* __restrict__ out) {
  __shared__ float s1[DM], s2[DM], sv[DM];
  int b = blockIdx.x, t = threadIdx.x;
  const float* row = h + ((size_t)b * LL + (LL - 1)) * DM;
  float v = row[t];
  s1[t] = v; s2[t] = v * v;
  __syncthreads();
  for (int st = DM / 2; st > 0; st >>= 1) {
    if (t < st) { s1[t] += s1[t + st]; s2[t] += s2[t + st]; }
    __syncthreads();
  }
  float mu  = s1[0] * (1.f / DM);
  float var = s2[0] * (1.f / DM) - mu * mu;
  sv[t] = (v - mu) * rsqrtf(var + 1e-5f) * g[t] + bta[t];
  __syncthreads();
  if (t < OUT_PER_B) {
    float acc = hb[t];
    const float* wr = hw + t * DM;
    #pragma unroll 8
    for (int d2 = 0; d2 < DM; d2++) acc = __builtin_fmaf(sv[d2], wr[d2], acc);
    out[b * OUT_PER_B + t] = acc;
  }
}

extern "C" void kernel_launch(void* const* d_in, const int* in_sizes, int n_in,
                              void* d_out, int out_size, void* d_ws, size_t ws_size,
                              hipStream_t stream) {
  const float* x      = (const float*)d_in[0];
  const float* enc_w  = (const float*)d_in[1];
  const float* enc_b  = (const float*)d_in[2];
  const float* in_w   = (const float*)d_in[3];
  const float* conv_w = (const float*)d_in[4];
  const float* conv_b = (const float*)d_in[5];
  const float* xp_w   = (const float*)d_in[6];
  const float* dt_w   = (const float*)d_in[7];
  const float* dt_b   = (const float*)d_in[8];
  const float* A_log  = (const float*)d_in[9];
  const float* Dp     = (const float*)d_in[10];
  const float* out_w  = (const float*)d_in[11];
  const float* ln_g   = (const float*)d_in[12];
  const float* ln_b   = (const float*)d_in[13];
  const float* head_w = (const float*)d_in[14];
  const float* head_b = (const float*)d_in[15];
  float* out = (float*)d_out;

  // -------- workspace carve-up (all offsets 256B aligned) --------
  char* wsp = (char*)d_ws;
  auto alloc = [&](size_t bytes) {
    char* p = wsp;
    wsp += (bytes + 255) & ~(size_t)255;
    return p;
  };
  float* h    = (float*)alloc((size_t)M_ROWS * DM * 4);        // residual stream
  u16*   bfA  = (u16*)  alloc((size_t)M_ROWS * DI * 2);        // bf16 A operand (h or xi)
  float* xz   = (float*)alloc((size_t)M_ROWS * 2 * DI * 4);    // in_proj output
  float* xi   = (float*)alloc((size_t)M_ROWS * DI * 4);        // conv+silu output
  float* dbl  = (float*)alloc((size_t)M_ROWS * NPROJ * 4);     // x_proj output
  float* dtv  = (float*)alloc((size_t)M_ROWS * DI * 4);        // softplus(dt)
  u16*   ybf  = (u16*)  alloc((size_t)M_ROWS * DI * 2);        // gated scan output (bf16)
  u16*   xbf  = (u16*)  alloc((size_t)M_ROWS * IN_DIM * 2);    // input in bf16
  u16*   wenc = (u16*)  alloc((size_t)DM * IN_DIM * 2);
  u16*   win  = (u16*)  alloc((size_t)NLAYERS * 2 * DI * DM * 2);
  u16*   wxp  = (u16*)  alloc((size_t)NLAYERS * NPROJ * DI * 2);
  u16*   wout = (u16*)  alloc((size_t)NLAYERS * DM * DI * 2);

  auto cvt = [&](const float* s, u16* d, int n) {
    k_f32_to_bf16<<<(n + 255) / 256, 256, 0, stream>>>(s, d, n);
  };
  auto gemm = [&](const u16* A, const u16* W, const float* bias, float* C,
                  int M, int N, int K, int accum) {
    int waveTiles = (M / 64) * (N / 16);     // 64x16 strip per wave
    k_gemm_bf16<<<(waveTiles + 7) / 8, 256, 0, stream>>>(A, W, bias, C, M, N, K, accum);
  };

  // -------- one-time conversions --------
  cvt(x,     xbf,  M_ROWS * IN_DIM);
  cvt(enc_w, wenc, DM * IN_DIM);
  cvt(in_w,  win,  NLAYERS * 2 * DI * DM);
  cvt(xp_w,  wxp,  NLAYERS * NPROJ * DI);
  cvt(out_w, wout, NLAYERS * DM * DI);

  // -------- encoder: h = x @ enc_w^T + enc_b --------
  gemm(xbf, wenc, enc_b, h, M_ROWS, DM, IN_DIM, /*accum=*/0);

  const int nPW = M_ROWS * DI;          // pointwise work items
  for (int lay = 0; lay < NLAYERS; lay++) {
    const u16* win_l  = win  + (size_t)lay * 2 * DI * DM;
    const u16* wxp_l  = wxp  + (size_t)lay * NPROJ * DI;
    const u16* wout_l = wout + (size_t)lay * DM * DI;

    // h -> bf16 A operand
    cvt(h, bfA, M_ROWS * DM);
    // xz = h @ in_proj_w^T   [M x 1024]
    gemm(bfA, win_l, nullptr, xz, M_ROWS, 2 * DI, DM, 0);
    // xi = silu(conv(xz[:, :512]))  (also writes bf16 copy into bfA)
    k_conv_silu<<<nPW / 256, 256, 0, stream>>>(
        xz, conv_w + (size_t)lay * DI * 4, conv_b + (size_t)lay * DI, xi, bfA);
    // dbl = xi @ xproj_w^T   [M x 48]
    gemm(bfA, wxp_l, nullptr, dbl, M_ROWS, NPROJ, DI, 0);
    // dt = softplus(dbl[:, :16] @ dt_w^T + dt_b)   (f32 VALU, K=16)
    k_dtproj<<<nPW / 256, 256, 0, stream>>>(
        dbl, dt_w + (size_t)lay * DI * DTR, dt_b + (size_t)lay * DI, dtv);
    // selective scan + Dp skip + silu(res) gate -> ybf (bf16)
    k_scan<<<16, 128, 0, stream>>>(
        dbl, dtv, xi, xz, A_log + (size_t)lay * DI * DST, Dp + (size_t)lay * DI, ybf);
    // h += y @ out_proj_w^T
    gemm(ybf, wout_l, nullptr, h, M_ROWS, DM, DI, /*accum=*/1);
  }

  // -------- LayerNorm on last token + head --------
  k_ln_head<<<BB, DM, 0, stream>>>(h, ln_g, ln_b, head_w, head_b, out);
}
